// QuantumLayer_62998580297887
// MI455X (gfx1250) — compile-verified, matrix operand order checked by reading, not run on hardware
//
#include <hip/hip_runtime.h>

typedef __attribute__((ext_vector_type(16))) __bf16 v16bf;
typedef __attribute__((ext_vector_type(8)))  __bf16 v8bf;
typedef __attribute__((ext_vector_type(8)))  float  v8f;

// ---------------- GEMM tiling constants ----------------
constexpr int BM  = 128;   // block tile M
constexpr int BN  = 128;   // block tile N
constexpr int BK  = 32;    // K per WMMA step
constexpr int LDT = 40;    // LDS row stride in bf16 (80B: 16B aligned, conflict-friendly)

// Load a [128 x 32] row-major tile (ld = K) into LDS as bf16, converting if fp32.
template<typename TA>
__device__ inline void load_rowmajor_tile(const TA* __restrict__ g, int ld,
                                          int row0, int k0,
                                          __bf16* __restrict__ lds, int tid) {
  const int r    = tid >> 1;      // 0..127
  const int half = tid & 1;       // 0..1 -> 16 elements each
  const TA* src = g + (size_t)(row0 + r) * ld + (k0 + half * 16);
  v8bf lo, hi;
  if constexpr (sizeof(TA) == 4) {
    const float4* s4 = (const float4*)src;
    float4 a = s4[0], b = s4[1], c = s4[2], d = s4[3];
    lo[0]=(__bf16)a.x; lo[1]=(__bf16)a.y; lo[2]=(__bf16)a.z; lo[3]=(__bf16)a.w;
    lo[4]=(__bf16)b.x; lo[5]=(__bf16)b.y; lo[6]=(__bf16)b.z; lo[7]=(__bf16)b.w;
    hi[0]=(__bf16)c.x; hi[1]=(__bf16)c.y; hi[2]=(__bf16)c.z; hi[3]=(__bf16)c.w;
    hi[4]=(__bf16)d.x; hi[5]=(__bf16)d.y; hi[6]=(__bf16)d.z; hi[7]=(__bf16)d.w;
  } else {
    const v8bf* s8 = (const v8bf*)src;
    lo = s8[0]; hi = s8[1];
  }
  v8bf* dst = (v8bf*)(lds + (size_t)r * LDT + half * 16);
  dst[0] = lo; dst[1] = hi;
}

// Load a [32 x 128] slice of a row-major [K x N] matrix, transposed into LDS as [n][k] bf16.
__device__ inline void load_kn_tile(const float* __restrict__ g, int ldn,
                                    int k0, int n0,
                                    __bf16* __restrict__ lds, int tid) {
  const int n  = tid & 127;       // column within tile (coalesced across lanes)
  const int kh = tid >> 7;        // 0..1 -> 16 k values each
  const float* base = g + (size_t)(k0 + kh * 16) * ldn + (n0 + n);
  v8bf lo, hi;
#pragma unroll
  for (int e = 0; e < 8; ++e) lo[e] = (__bf16)base[(size_t)e * ldn];
#pragma unroll
  for (int e = 0; e < 8; ++e) hi[e] = (__bf16)base[(size_t)(e + 8) * ldn];
  v8bf* dst = (v8bf*)(lds + (size_t)n * LDT + kh * 16);
  dst[0] = lo; dst[1] = hi;
}

// C[M,N] = A[M,K] * B + bias.  B is [N][K] row-major if !B_IS_KN (i.e. Bmat = stored^T),
// or [K][N] row-major if B_IS_KN.  fp32 accumulate via v_wmma_f32_16x16x32_bf16.
template<typename TA, bool B_IS_KN>
__global__ __launch_bounds__(256) void gemm_bf16_kernel(
    const TA* __restrict__ A, const float* __restrict__ Bm,
    const float* __restrict__ bias, float* __restrict__ C,
    int M, int N, int K) {
  __shared__ __bf16 As[BM * LDT];
  __shared__ __bf16 Bs[BN * LDT];

  const int tid  = threadIdx.x;
  const int lane = tid & 31;
  const int wave = tid >> 5;
  const int wm   = wave & 3;      // 4 wave-rows of 32
  const int wn   = wave >> 2;     // 2 wave-cols of 64
  const int m0   = blockIdx.x * BM;
  const int n0   = blockIdx.y * BN;
  const int r    = lane & 15;
  const int g    = lane >> 4;

  v8f acc[2][4];
#pragma unroll
  for (int i = 0; i < 2; ++i)
#pragma unroll
    for (int j = 0; j < 4; ++j)
#pragma unroll
      for (int e = 0; e < 8; ++e) acc[i][j][e] = 0.0f;

  for (int k0 = 0; k0 < K; k0 += BK) {
    load_rowmajor_tile<TA>(A, K, m0, k0, As, tid);
    if constexpr (B_IS_KN) load_kn_tile(Bm, N, k0, n0, Bs, tid);
    else                   load_rowmajor_tile<float>(Bm, K, n0, k0, Bs, tid);
    __syncthreads();

    if (k0 + BK < K) {   // hint next K-tile into cache (global_prefetch_b8)
      __builtin_prefetch(A + (size_t)(m0 + (tid >> 1)) * K + (k0 + BK), 0, 0);
      if constexpr (B_IS_KN)
        __builtin_prefetch(Bm + (size_t)(k0 + BK + (tid >> 7) * 16) * N + n0 + (tid & 127), 0, 0);
      else
        __builtin_prefetch(Bm + (size_t)(n0 + (tid >> 1)) * K + (k0 + BK), 0, 0);
    }

    // Fragment gathers per documented 16-bit layouts:
    //   vector elems 0..7  <-> K = 8g .. 8g+7
    //   vector elems 8..15 <-> K = 16+8g .. 16+8g+7
    v16bf af[2], bfr[4];
#pragma unroll
    for (int i = 0; i < 2; ++i) {
      const __bf16* p = As + (size_t)(wm * 32 + i * 16 + r) * LDT + 8 * g;
      v8bf lo = *(const v8bf*)p;
      v8bf hi = *(const v8bf*)(p + 16);
      af[i] = __builtin_shufflevector(lo, hi, 0,1,2,3,4,5,6,7,8,9,10,11,12,13,14,15);
    }
#pragma unroll
    for (int j = 0; j < 4; ++j) {
      const __bf16* p = Bs + (size_t)(wn * 64 + j * 16 + r) * LDT + 8 * g;
      v8bf lo = *(const v8bf*)p;
      v8bf hi = *(const v8bf*)(p + 16);
      bfr[j] = __builtin_shufflevector(lo, hi, 0,1,2,3,4,5,6,7,8,9,10,11,12,13,14,15);
    }

#pragma unroll
    for (int i = 0; i < 2; ++i)
#pragma unroll
      for (int j = 0; j < 4; ++j)
        acc[i][j] = __builtin_amdgcn_wmma_f32_16x16x32_bf16(
            false, af[i], false, bfr[j], (short)0, acc[i][j], false, false);
    __syncthreads();
  }

  // Epilogue: C/D layout -> row = vi + 8g, col = lane&15
#pragma unroll
  for (int i = 0; i < 2; ++i) {
#pragma unroll
    for (int j = 0; j < 4; ++j) {
      const int col = n0 + wn * 64 + j * 16 + r;
      const float bv = bias ? bias[col] : 0.0f;
#pragma unroll
      for (int vi = 0; vi < 8; ++vi) {
        const int row = m0 + wm * 32 + i * 16 + g * 8 + vi;
        C[(size_t)row * N + col] = acc[i][j][vi] + bv;
      }
    }
  }
}

// mult[q,h] = prod_{j>q} ent[q,j,h] * prod_{i<q} ent[i,q,h]
__global__ void mult_kernel(const float* __restrict__ ent, float* __restrict__ mult,
                            int Q, int H) {
  int idx = blockIdx.x * blockDim.x + threadIdx.x;
  if (idx >= Q * H) return;
  int q = idx / H, h = idx - q * H;
  float p = 1.0f;
  for (int j = q + 1; j < Q; ++j) p *= ent[((size_t)q * Q + j) * H + h];
  for (int i = 0; i < q; ++i)     p *= ent[((size_t)i * Q + q) * H + h];
  mult[idx] = p;
}

// Per-row L2 normalize with F.normalize eps; optionally emit fp32 and/or bf16.
__global__ __launch_bounds__(256) void rownorm_kernel(
    const float* __restrict__ Cin, float* __restrict__ outF,
    __bf16* __restrict__ outB, int Ncols) {
  __shared__ float red[256];
  const int b = blockIdx.x, t = threadIdx.x;
  const float* row = Cin + (size_t)b * Ncols;
  float ss = 0.0f;
  for (int h = t; h < Ncols; h += 256) { float v = row[h]; ss += v * v; }
  red[t] = ss; __syncthreads();
  for (int s = 128; s > 0; s >>= 1) {
    if (t < s) red[t] += red[t + s];
    __syncthreads();
  }
  const float inv = 1.0f / fmaxf(sqrtf(red[0]), 1e-12f);
  for (int h = t; h < Ncols; h += 256) {
    float v = row[h] * inv;
    if (outF) outF[(size_t)b * Ncols + h] = v;
    if (outB) outB[(size_t)b * Ncols + h] = (__bf16)v;
  }
}

// Per row b: n[q]=||s*mult[q]||, then w[h] = s[h]*(1/Q)*sum_q mult[q,h]/max(n[q],eps)
__global__ __launch_bounds__(256) void collapse_kernel(
    const float* __restrict__ s, const float* __restrict__ mult,
    __bf16* __restrict__ wbf, int Q, int H) {
  __shared__ float sh_s[2048];
  __shared__ float red[256];
  __shared__ float invn[32];
  const int b = blockIdx.x, t = threadIdx.x;
  const float* srow = s + (size_t)b * H;
  for (int h = t; h < H; h += 256) sh_s[h] = srow[h];
  __syncthreads();
  for (int q = 0; q < Q; ++q) {
    float ss = 0.0f;
    for (int h = t; h < H; h += 256) {
      float v = sh_s[h] * mult[(size_t)q * H + h];
      ss += v * v;
    }
    red[t] = ss; __syncthreads();
    for (int st = 128; st > 0; st >>= 1) {
      if (t < st) red[t] += red[t + st];
      __syncthreads();
    }
    if (t == 0) invn[q] = 1.0f / fmaxf(sqrtf(red[0]), 1e-12f);
    __syncthreads();
  }
  const float iQ = 1.0f / (float)Q;
  for (int h = t; h < H; h += 256) {
    float a = 0.0f;
    for (int q = 0; q < Q; ++q) a += invn[q] * mult[(size_t)q * H + h];
    wbf[(size_t)b * H + h] = (__bf16)(sh_s[h] * a * iQ);
  }
}

extern "C" void kernel_launch(void* const* d_in, const int* in_sizes, int n_in,
                              void* d_out, int out_size, void* d_ws, size_t ws_size,
                              hipStream_t stream) {
  const float* x      = (const float*)d_in[0];
  const float* proj_w = (const float*)d_in[1];  // [H,I]
  const float* proj_b = (const float*)d_in[2];  // [H]
  const float* sup_w  = (const float*)d_in[3];  // [H,H]
  const float* ent_w  = (const float*)d_in[4];  // [Q,Q,H]
  const float* meas_w = (const float*)d_in[5];  // [H,H]

  const int H = in_sizes[2];
  const int I = in_sizes[1] / H;
  const int B = in_sizes[0] / I;
  int Q = 1; while (Q * Q * H < in_sizes[4]) ++Q;

  const size_t szBH4 = (size_t)B * H * sizeof(float);
  const size_t szBH2 = (size_t)B * H * sizeof(__bf16);
  char* w = (char*)d_ws;
  float*  C1   = (float*)w;                              // GEMM1 / GEMM3 raw out
  float*  C2   = (float*)(w + szBH4);                    // GEMM2 raw out -> s (in place)
  __bf16* hbf  = (__bf16*)(w + 2 * szBH4);               // normalized h, bf16
  __bf16* wbf  = (__bf16*)(w + 2 * szBH4 + szBH2);       // collapsed mean, bf16
  float*  mult = (float*)(w + 2 * szBH4 + 2 * szBH2);    // [Q,H]

  const dim3 blk(256);
  const dim3 gg(B / BM, H / BN);

  // entanglement multiplier (B-independent)
  mult_kernel<<<(Q * H + 255) / 256, blk, 0, stream>>>(ent_w, mult, Q, H);

  // h = l2norm(x @ proj_w.T + b)  : B matrix is proj_w stored [N][K]
  gemm_bf16_kernel<float, false><<<gg, blk, 0, stream>>>(x, proj_w, proj_b, C1, B, H, I);
  rownorm_kernel<<<B, blk, 0, stream>>>(C1, nullptr, hbf, H);

  // s = l2norm(h @ sup_w)         : B matrix stored [K][N]
  gemm_bf16_kernel<__bf16, true><<<gg, blk, 0, stream>>>(hbf, sup_w, nullptr, C2, B, H, H);
  rownorm_kernel<<<B, blk, 0, stream>>>(C2, C2, nullptr, H);

  // mean over qubits of l2norm(s * mult[q])  ->  bf16 input of GEMM3
  collapse_kernel<<<B, blk, 0, stream>>>(C2, mult, wbf, Q, H);

  // out = l2norm(mean @ meas_w)
  gemm_bf16_kernel<__bf16, true><<<gg, blk, 0, stream>>>(wbf, meas_w, nullptr, C1, B, H, H);
  rownorm_kernel<<<B, blk, 0, stream>>>(C1, (float*)d_out, nullptr, H);
}
